// BDH_5420248728257
// MI455X (gfx1250) — compile-verified, matrix operand order checked by reading, not run on hardware
//
#include <hip/hip_runtime.h>

typedef __bf16 bf16_t;
typedef __attribute__((ext_vector_type(16))) __bf16 v16bf;
typedef __attribute__((ext_vector_type(8)))  __bf16 v8bf;
typedef __attribute__((ext_vector_type(8)))  float  v8f;
typedef __attribute__((ext_vector_type(4)))  int    v4i;

#define TT   1024
#define EE   768
#define HH   12
#define NNI  2048
#define VV   32000
#define NLAYER 6

#define KSLAB 64
#define LDW   72   /* 64 + 8 pad: 144B row stride = 36 banks, conflict-free */

// CDNA5 async global->LDS path (ASYNCcnt), if this toolchain exposes it.
#if defined(__has_builtin)
#if __has_builtin(__builtin_amdgcn_global_load_async_to_lds_b128)
#define HAVE_ASYNC_LDS 1
#endif
#endif
#ifndef HAVE_ASYNC_LDS
#define HAVE_ASYNC_LDS 0
#endif

#if HAVE_ASYNC_LDS
typedef __attribute__((address_space(1))) v4i gas_v4i;
typedef __attribute__((address_space(3))) v4i las_v4i;
static __device__ __forceinline__ void async_cp16(const bf16_t* g, bf16_t* l) {
  // (global v4i*, shared v4i*, imm offset, imm cpol)
  __builtin_amdgcn_global_load_async_to_lds_b128(
      (gas_v4i*)(bf16_t*)g, (las_v4i*)l, 0, 0);
}
static __device__ __forceinline__ void wait_async0() {
#if __has_builtin(__builtin_amdgcn_s_wait_asynccnt)
  __builtin_amdgcn_s_wait_asynccnt(0);
#else
  asm volatile("s_wait_asynccnt 0x0" ::: "memory");
#endif
}
#endif

// ---------------------------------------------------------------------------
// WMMA helpers
// ---------------------------------------------------------------------------
static __device__ __forceinline__ v8f wmma_bf16(v16bf a, v16bf b, v8f c) {
  return __builtin_amdgcn_wmma_f32_16x16x32_bf16(false, a, false, b,
                                                 (short)0, c, false, false);
}

// 16x32 bf16 fragment from an LDS tile at k-offset kk within the slab.
// lanes 0-15: K={0..7,16..23}; lanes 16-31: K={8..15,24..31}; row = lane&15.
static __device__ __forceinline__ v16bf frag_lds(const bf16_t tile[][LDW],
                                                 int row, int kk) {
  const int lane = threadIdx.x & 31;
  const bf16_t* q = &tile[row + (lane & 15)][kk + ((lane & 16) >> 1)];
  v8bf lo = *(const v8bf*)(q);
  v8bf hi = *(const v8bf*)(q + 16);
  v16bf f;
#pragma unroll
  for (int i = 0; i < 8; ++i) { f[i] = lo[i]; f[i + 8] = hi[i]; }
  return f;
}

// ---------------------------------------------------------------------------
// Block reductions (wave32)
// ---------------------------------------------------------------------------
static __device__ __forceinline__ float wave_sum(float v) {
#pragma unroll
  for (int o = 16; o > 0; o >>= 1) v += __shfl_down(v, o, 32);
  return v;
}
static __device__ __forceinline__ float wave_max(float v) {
#pragma unroll
  for (int o = 16; o > 0; o >>= 1) v = fmaxf(v, __shfl_down(v, o, 32));
  return v;
}
static __device__ __forceinline__ float block_sum(float v) {
  __shared__ float s[8];
  const int lane = threadIdx.x & 31, wid = threadIdx.x >> 5;
  v = wave_sum(v);
  if (lane == 0) s[wid] = v;
  __syncthreads();
  if (threadIdx.x == 0) {
    float r = 0.f;
#pragma unroll
    for (int i = 0; i < 8; ++i) r += s[i];
    s[0] = r;
  }
  __syncthreads();
  float r = s[0];
  __syncthreads();
  return r;
}
static __device__ __forceinline__ float block_max(float v) {
  __shared__ float s[8];
  const int lane = threadIdx.x & 31, wid = threadIdx.x >> 5;
  v = wave_max(v);
  if (lane == 0) s[wid] = v;
  __syncthreads();
  if (threadIdx.x == 0) {
    float r = -3e38f;
#pragma unroll
    for (int i = 0; i < 8; ++i) r = fmaxf(r, s[i]);
    s[0] = r;
  }
  __syncthreads();
  float r = s[0];
  __syncthreads();
  return r;
}

// ---------------------------------------------------------------------------
// LDS-staged NT GEMM:  C[M,N] = sum_rep A_rep[M,K] * (Bt_rep[N,K])^T
//  - 256 threads = 8 waves; block tile 128(M) x 128(N); k-slab 64
//  - waves 4(M) x 2(N): each wave 32x64 -> 16 wmma per slab (2 k-steps x 2x4)
//  - A/B slabs staged in LDS; async global->LDS double-buffer when available
// EPI: 0 = f32; 1 = relu->bf16; 2 = *scale + causal mask -> f32;
//      3 = relu * aux(bf16) -> bf16
// ---------------------------------------------------------------------------
template <int EPI>
__global__ __launch_bounds__(256) void gemm_nt(
    const bf16_t* __restrict__ A, const bf16_t* __restrict__ Bt,
    void* __restrict__ Cout, const bf16_t* __restrict__ aux,
    int M, int N, int K, int nrep,
    long sAz, long sBz, long sCz, long sXz,
    long sAh, long sBh, float scale)
{
  if (EPI == 2) {
    // 128x128 tile entirely above the causal diagonal: never read -> skip
    if ((int)blockIdx.x > (int)blockIdx.y) return;
  }

#if HAVE_ASYNC_LDS
  __shared__ bf16_t As[2][128][LDW];
  __shared__ bf16_t Bs[2][128][LDW];
#else
  __shared__ bf16_t As[1][128][LDW];
  __shared__ bf16_t Bs[1][128][LDW];
#endif

  const int tid  = threadIdx.x;
  const int lane = tid & 31;
  const int w    = tid >> 5;
  const int waveM = w >> 1;            // 0..3 -> 32 rows each
  const int waveN = w & 1;             // 0..1 -> 64 cols each

  const bf16_t* Ab = A  + (long)blockIdx.z * sAz + (long)blockIdx.y * 128 * K;
  const bf16_t* Bb = Bt + (long)blockIdx.z * sBz + (long)blockIdx.x * 128 * K;

  const int fr = tid >> 3;             // 0..31 (fill rows: fr, fr+32, fr+64, fr+96)
  const int fk = (tid & 7) << 3;       // 0..56 (fill k segment)

  const int kpr = K / KSLAB;           // slabs per rep
  const int S   = nrep * kpr;

  v8f acc[8];
#pragma unroll
  for (int i = 0; i < 8; ++i) acc[i] = (v8f){0.f,0.f,0.f,0.f,0.f,0.f,0.f,0.f};

  auto gsrc = [&](const bf16_t* base, long sh, int s, int r) -> const bf16_t* {
    const int rep = s / kpr;
    const int k0  = (s - rep * kpr) * KSLAB;
    return base + (long)rep * sh + (long)r * K + k0 + fk;
  };

  auto compute = [&](int buf) {
#pragma unroll
    for (int kk = 0; kk < KSLAB; kk += 32) {
      v16bf a0 = frag_lds(As[buf], waveM * 32,      kk);
      v16bf a1 = frag_lds(As[buf], waveM * 32 + 16, kk);
      v16bf b0 = frag_lds(Bs[buf], waveN * 64,      kk);
      v16bf b1 = frag_lds(Bs[buf], waveN * 64 + 16, kk);
      v16bf b2 = frag_lds(Bs[buf], waveN * 64 + 32, kk);
      v16bf b3 = frag_lds(Bs[buf], waveN * 64 + 48, kk);
      acc[0] = wmma_bf16(a0, b0, acc[0]);
      acc[1] = wmma_bf16(a0, b1, acc[1]);
      acc[2] = wmma_bf16(a0, b2, acc[2]);
      acc[3] = wmma_bf16(a0, b3, acc[3]);
      acc[4] = wmma_bf16(a1, b0, acc[4]);
      acc[5] = wmma_bf16(a1, b1, acc[5]);
      acc[6] = wmma_bf16(a1, b2, acc[6]);
      acc[7] = wmma_bf16(a1, b3, acc[7]);
    }
  };

#if HAVE_ASYNC_LDS
  auto fill = [&](int buf, int s) {
#pragma unroll
    for (int j = 0; j < 4; ++j) {
      const int r = fr + j * 32;
      async_cp16(gsrc(Ab, sAh, s, r), &As[buf][r][fk]);
      async_cp16(gsrc(Bb, sBh, s, r), &Bs[buf][r][fk]);
    }
  };
  fill(0, 0);
  wait_async0();
  __syncthreads();
  int buf = 0;
  for (int s = 0; s < S; ++s) {
    if (s + 1 < S) fill(buf ^ 1, s + 1);   // overlap next slab with compute
    compute(buf);
    wait_async0();                          // my async writes visible
    __syncthreads();                        // everyone's writes visible
    buf ^= 1;
  }
#else
  for (int s = 0; s < S; ++s) {
    v8bf ra[4], rb[4];
#pragma unroll
    for (int j = 0; j < 4; ++j) {
      ra[j] = *(const v8bf*)gsrc(Ab, sAh, s, fr + j * 32);
      rb[j] = *(const v8bf*)gsrc(Bb, sBh, s, fr + j * 32);
    }
    __syncthreads();                        // prior reads of LDS done
#pragma unroll
    for (int j = 0; j < 4; ++j) {
      *(v8bf*)&As[0][fr + j * 32][fk] = ra[j];
      *(v8bf*)&Bs[0][fr + j * 32][fk] = rb[j];
    }
    __syncthreads();
    compute(0);
  }
#endif

  // Epilogue. C/D layout: n = lane&15, m = 8*(lane>>4) + vgpr-index.
  const int nn = lane & 15;
  const int mo = (lane >> 4) << 3;
  const int mbase = blockIdx.y * 128 + waveM * 32;
  const int nbase = blockIdx.x * 128 + waveN * 64;
#pragma unroll
  for (int t = 0; t < 8; ++t) {
    const int mi = t >> 2, ni = t & 3;
#pragma unroll
    for (int i = 0; i < 8; ++i) {
      const int m = mbase + mi * 16 + mo + i;
      const int n = nbase + ni * 16 + nn;
      const long ci = (long)blockIdx.z * sCz + (long)m * N + n;
      float v = acc[t][i];
      if (EPI == 0) {
        ((float*)Cout)[ci] = v;
      } else if (EPI == 1) {
        ((bf16_t*)Cout)[ci] = (bf16_t)fmaxf(v, 0.f);
      } else if (EPI == 2) {
        v *= scale;
        if (n > m) v = -1e30f;
        ((float*)Cout)[ci] = v;
      } else {
        const float a = (float)aux[(long)blockIdx.z * sXz + (long)m * N + n];
        ((bf16_t*)Cout)[ci] = (bf16_t)(fmaxf(v, 0.f) * a);
      }
    }
  }
}

// ---------------------------------------------------------------------------
// Pointwise / LN / softmax kernels
// ---------------------------------------------------------------------------
__global__ __launch_bounds__(256) void cvt_t(const float* __restrict__ in,
                                             bf16_t* __restrict__ out,
                                             int H, int R, int C) {
  const long i = (long)blockIdx.x * 256 + threadIdx.x;
  if (i >= (long)H * R * C) return;
  const int  c = (int)(i % C);
  const long t = i / C;
  const int  r = (int)(t % R);
  const int  h = (int)(t / R);
  out[((long)h * C + c) * R + r] = (bf16_t)in[i];   // [H,R,C] -> [H,C,R]
}

__global__ __launch_bounds__(256) void cvt(const float* __restrict__ in,
                                           bf16_t* __restrict__ out, long n) {
  const long i = (long)blockIdx.x * 256 + threadIdx.x;
  if (i < n) out[i] = (bf16_t)in[i];
}

__global__ __launch_bounds__(256) void embed_ln(
    const int* __restrict__ idx, const float* __restrict__ embed,
    const float* __restrict__ lnw, const float* __restrict__ lnb,
    float* __restrict__ x, bf16_t* __restrict__ xbf, bf16_t* __restrict__ xTbf)
{
  const int t = blockIdx.x;
  const float* row = embed + (long)idx[t] * EE;
  float s = 0.f, s2 = 0.f;
  for (int e = threadIdx.x; e < EE; e += 256) { float v = row[e]; s += v; s2 += v * v; }
  s = block_sum(s); s2 = block_sum(s2);
  const float m  = s / EE;
  const float rs = rsqrtf(fmaxf(s2 / EE - m * m, 0.f) + 1e-5f);
  for (int e = threadIdx.x; e < EE; e += 256) {
    const float v = (row[e] - m) * rs * lnw[e] + lnb[e];
    x[(long)t * EE + e] = v;
    const bf16_t bv = (bf16_t)v;
    xbf[(long)t * EE + e] = bv;
    xTbf[(long)e * TT + t] = bv;
  }
}

__global__ __launch_bounds__(256) void ln_rows(
    const float* __restrict__ in, bf16_t* __restrict__ out,
    const float* __restrict__ lnw, const float* __restrict__ lnb)
{
  const long r = (long)blockIdx.y * TT + blockIdx.x;   // (h, t)
  const float* row = in + r * EE;
  bf16_t* orow = out + r * EE;
  float s = 0.f, s2 = 0.f;
  for (int e = threadIdx.x; e < EE; e += 256) { float v = row[e]; s += v; s2 += v * v; }
  s = block_sum(s); s2 = block_sum(s2);
  const float m  = s / EE;
  const float rs = rsqrtf(fmaxf(s2 / EE - m * m, 0.f) + 1e-5f);
  for (int e = threadIdx.x; e < EE; e += 256)
    orow[e] = (bf16_t)((row[e] - m) * rs * lnw[e] + lnb[e]);
}

__global__ __launch_bounds__(256) void residual_ln(
    const float* __restrict__ ymlp, float* __restrict__ x,
    bf16_t* __restrict__ xbf, bf16_t* __restrict__ xTbf,
    const float* __restrict__ lnw, const float* __restrict__ lnb)
{
  const int t = blockIdx.x;
  const float* yr = ymlp + (long)t * EE;
  float* xr = x + (long)t * EE;
  float s = 0.f, s2 = 0.f;
  for (int e = threadIdx.x; e < EE; e += 256) { float v = yr[e]; s += v; s2 += v * v; }
  s = block_sum(s); s2 = block_sum(s2);
  const float m1  = s / EE;
  const float rs1 = rsqrtf(fmaxf(s2 / EE - m1 * m1, 0.f) + 1e-5f);
  float z[4];
  float zs = 0.f, zs2 = 0.f;
  for (int e = threadIdx.x, j = 0; e < EE; e += 256, ++j) {
    const float v = xr[e] + (yr[e] - m1) * rs1 * lnw[e] + lnb[e];
    z[j] = v; zs += v; zs2 += v * v;
  }
  zs = block_sum(zs); zs2 = block_sum(zs2);
  const float m2  = zs / EE;
  const float rs2 = rsqrtf(fmaxf(zs2 / EE - m2 * m2, 0.f) + 1e-5f);
  for (int e = threadIdx.x, j = 0; e < EE; e += 256, ++j) {
    const float v = (z[j] - m2) * rs2 * lnw[e] + lnb[e];
    xr[e] = v;
    const bf16_t bv = (bf16_t)v;
    xbf[(long)t * EE + e] = bv;
    xTbf[(long)e * TT + t] = bv;
  }
}

__global__ __launch_bounds__(256) void softmax_causal(
    const float* __restrict__ scores, bf16_t* __restrict__ attn)
{
  const int q = blockIdx.x, h = blockIdx.y;
  const float* row = scores + ((long)h * TT + q) * TT;
  bf16_t* out = attn + ((long)h * TT + q) * TT;
  const int len = q + 1;
  float mx = -3e38f;
  for (int k = threadIdx.x; k < len; k += 256) mx = fmaxf(mx, row[k]);
  mx = block_max(mx);
  float s = 0.f;
  for (int k = threadIdx.x; k < len; k += 256) s += __expf(row[k] - mx);
  s = block_sum(s);
  const float inv = 1.f / s;
  for (int k = threadIdx.x; k < TT; k += 256)
    out[k] = (k < len) ? (bf16_t)(__expf(row[k] - mx) * inv) : (bf16_t)0.f;
}

// ---------------------------------------------------------------------------
// Launch
// ---------------------------------------------------------------------------
extern "C" void kernel_launch(void* const* d_in, const int* in_sizes, int n_in,
                              void* d_out, int out_size, void* d_ws, size_t ws_size,
                              hipStream_t stream) {
  const int*   idx   = (const int*)  d_in[0];
  const float* embed = (const float*)d_in[1];
  const float* enc   = (const float*)d_in[2];
  const float* encv  = (const float*)d_in[3];
  const float* dec   = (const float*)d_in[4];
  const float* lmh   = (const float*)d_in[5];
  const float* lnw   = (const float*)d_in[6];
  const float* lnb   = (const float*)d_in[7];
  float* out = (float*)d_out;

  char* p = (char*)d_ws;
  auto carve = [&](size_t bytes) -> void* {
    void* r = (void*)p; p += (bytes + 255) & ~(size_t)255; return r;
  };
  float*  x      = (float*) carve((size_t)TT * EE * 4);
  bf16_t* xbf    = (bf16_t*)carve((size_t)TT * EE * 2);
  bf16_t* xTbf   = (bf16_t*)carve((size_t)EE * TT * 2);
  bf16_t* encT   = (bf16_t*)carve((size_t)HH * NNI * EE * 2);   // [H,N,E] = B^T
  bf16_t* encvT  = (bf16_t*)carve((size_t)HH * NNI * EE * 2);   // [H,N,E]
  bf16_t* decT   = (bf16_t*)carve((size_t)HH * EE * NNI * 2);   // [H,E,N] = B^T
  bf16_t* lmbf   = (bf16_t*)carve((size_t)VV * EE * 2);         // [V,E] already B^T
  bf16_t* xs     = (bf16_t*)carve((size_t)HH * TT * NNI * 2);
  float*  scores = (float*) carve((size_t)HH * TT * TT * 4);
  bf16_t* attn   = (bf16_t*)carve((size_t)HH * TT * TT * 2);
  float*  y      = (float*) carve((size_t)HH * TT * EE * 4);
  bf16_t* ybf    = (bf16_t*)carve((size_t)HH * TT * EE * 2);
  bf16_t* xy     = (bf16_t*)carve((size_t)HH * TT * NNI * 2);
  float*  ymlp   = (float*) carve((size_t)TT * EE * 4);

  {
    const long n1 = (long)HH * EE * NNI;
    const unsigned g1 = (unsigned)((n1 + 255) / 256);
    cvt_t<<<dim3(g1), dim3(256), 0, stream>>>(enc,  encT,  HH, EE,  NNI);
    cvt_t<<<dim3(g1), dim3(256), 0, stream>>>(encv, encvT, HH, EE,  NNI);
    cvt_t<<<dim3(g1), dim3(256), 0, stream>>>(dec,  decT,  HH, NNI, EE);
    const long n2 = (long)VV * EE;
    cvt<<<dim3((unsigned)((n2 + 255) / 256)), dim3(256), 0, stream>>>(lmh, lmbf, n2);
  }

  embed_ln<<<dim3(TT), dim3(256), 0, stream>>>(idx, embed, lnw, lnb, x, xbf, xTbf);

  const float rscale = 1.f / sqrtf((float)NNI);

  for (int l = 0; l < NLAYER; ++l) {
    // x_sparse[h] = relu(x @ enc[h])            M=T, N=NNI, K=E
    gemm_nt<1><<<dim3(NNI / 128, TT / 128, HH), dim3(256), 0, stream>>>(
        xbf, encT, xs, nullptr, TT, NNI, EE, 1,
        0L, (long)NNI * EE, (long)TT * NNI, 0L, 0L, 0L, 1.f);
    // scores[h] = xs[h] @ xs[h]^T * scale (+causal mask)   M=T, N=T, K=NNI
    gemm_nt<2><<<dim3(TT / 128, TT / 128, HH), dim3(256), 0, stream>>>(
        xs, xs, scores, nullptr, TT, TT, NNI, 1,
        (long)TT * NNI, (long)TT * NNI, (long)TT * TT, 0L, 0L, 0L, rscale);
    softmax_causal<<<dim3(TT, HH), dim3(256), 0, stream>>>(scores, attn);
    // y[h] = attn[h] @ x                        M=T, N=E, K=T   (B^T = xT)
    gemm_nt<0><<<dim3(EE / 128, TT / 128, HH), dim3(256), 0, stream>>>(
        attn, xTbf, y, nullptr, TT, EE, TT, 1,
        (long)TT * TT, 0L, (long)TT * EE, 0L, 0L, 0L, 1.f);
    ln_rows<<<dim3(TT, HH), dim3(256), 0, stream>>>(y, ybf, lnw, lnb);
    // xy[h] = relu(y[h] @ encv[h]) * xs[h]      M=T, N=NNI, K=E
    gemm_nt<3><<<dim3(NNI / 128, TT / 128, HH), dim3(256), 0, stream>>>(
        ybf, encvT, xy, xs, TT, NNI, EE, 1,
        (long)TT * EE, (long)NNI * EE, (long)TT * NNI, (long)TT * NNI, 0L, 0L, 1.f);
    // ymlp = sum_h xy[h] @ dec[h]               M=T, N=E, K=NNI, nrep=H
    gemm_nt<0><<<dim3(EE / 128, TT / 128, 1), dim3(256), 0, stream>>>(
        xy, decT, ymlp, nullptr, TT, EE, NNI, HH,
        0L, 0L, 0L, 0L, (long)TT * NNI, (long)EE * NNI, 1.f);
    residual_ln<<<dim3(TT), dim3(256), 0, stream>>>(ymlp, x, xbf, xTbf, lnw, lnb);
  }

  // logits = x @ lm_head^T                      M=T, N=V, K=E
  gemm_nt<0><<<dim3(VV / 128, TT / 128, 1), dim3(256), 0, stream>>>(
      xbf, lmbf, out, nullptr, TT, VV, EE, 1,
      0L, 0L, 0L, 0L, 0L, 0L, 1.f);
}
